// IMF_88871463289192
// MI455X (gfx1250) — matrix operand order repaired and checked by reading, not compiled, not run on hardware
//
#include <hip/hip_runtime.h>
#include <hip/hip_bf16.h>

// ---------------------------------------------------------------------------
// IMF (integer matrix factorization) for MI455X / gfx1250, wave32 + WMMA.
//   per batch:  u,v in R^{512x32};  10 alternating Gauss-Seidel sweeps:
//     a = x@v, b = v^T v, per-row sweep of u;  then same with x^T and u for v.
//   output: u @ v^T.
// GEMMs: v_wmma_f32_16x16x32_f16, K unrolled x2 with independent accumulators
// (hides WMMA->WMMA RAW hazard), double-buffered per-wave LDS staging.
// Sweep: b (32x32 Gram) broadcast into LDS via the Tensor Data Mover.
// Problem is L2-resident (x = 64 MB << 192 MB L2); latency-bound, so the
// design is straight-line wave-level tiles and few wide kernels.
// ---------------------------------------------------------------------------

typedef __attribute__((ext_vector_type(16))) _Float16 v16h;
typedef __attribute__((ext_vector_type(8)))  _Float16 v8h;
typedef __attribute__((ext_vector_type(8)))  float    v8f;
typedef __attribute__((ext_vector_type(4))) unsigned int u32x4;
typedef __attribute__((ext_vector_type(8))) int          i32x8;
typedef __attribute__((ext_vector_type(4))) int          i32x4;

#define BATCH 64
#define NDIM  512
#define RANK  32
#define IMF_EPS 1e-8f
#define NUM_ITERS 10

// Batched WMMA GEMM:  C[M x Nn] = op(A)[M x K] * op(B)[K x Nn]
// TA/TB compile-time: straight-line staging, no divergence (EXEC all-1s).
// One wave owns one 16x16 C tile across the full K, 2 K-steps in flight.
template<int TA, int TB>
__global__ __launch_bounds__(128) void imf_gemm(
    const float* __restrict__ A,  int lda, long sA,
    const float* __restrict__ Bm, int ldb, long sB,
    float*       __restrict__ C,  int ldc, long sC,
    int M, int Nn, int K)
{
  __shared__ __align__(64) _Float16 ldsA[4][2][16 * 32]; // [m 0..15][k 0..31]
  __shared__ __align__(64) _Float16 ldsB[4][2][16 * 32]; // transposed: [n][k]

  const int lane = threadIdx.x & 31;
  const int warp = threadIdx.x >> 5;

  const int nTiles        = Nn >> 4;
  const int tilesPerBatch = (M >> 4) * nTiles;
  const int tile = blockIdx.x * 4 + warp;     // grid sized exactly: no guard
  const int b    = tile / tilesPerBatch;
  const int t    = tile - b * tilesPerBatch;
  const int mT   = t / nTiles;
  const int nT   = t - mT * nTiles;

  const float* Ab = A  + (size_t)b * sA;
  const float* Bb = Bm + (size_t)b * sB;
  float*       Cb = C  + (size_t)b * sC;

  _Float16* la0 = ldsA[warp][0]; _Float16* la1 = ldsA[warp][1];
  _Float16* lb0 = ldsB[warp][0]; _Float16* lb1 = ldsB[warp][1];

  // Per-lane slab pointers: each lane loads 16 contiguous floats per K-step.
  const int half = lane & 1;
  const int r2   = lane >> 1;
  const float* pA; size_t stepA;
  if (TA == 0) { pA = Ab + (size_t)(mT * 16 + r2) * lda + half * 16; stepA = 32; }
  else         { pA = Ab + (size_t)lane * lda + mT * 16;  stepA = (size_t)32 * lda; }
  const float* pB; size_t stepB;
  if (TB == 0) { pB = Bb + (size_t)lane * ldb + nT * 16;  stepB = (size_t)32 * ldb; }
  else         { pB = Bb + (size_t)(nT * 16 + r2) * ldb + half * 16; stepB = 32; }

  // ---- staging: 4x b128 global loads, cvt f16, LDS store (contig or scatter)
  auto stageA = [&](const float* p, _Float16* l) {
    float t4[16];
    *(float4*)&t4[0]  = ((const float4*)p)[0];
    *(float4*)&t4[4]  = ((const float4*)p)[1];
    *(float4*)&t4[8]  = ((const float4*)p)[2];
    *(float4*)&t4[12] = ((const float4*)p)[3];
    _Float16 h[16];
    #pragma unroll
    for (int i = 0; i < 16; ++i) h[i] = (_Float16)t4[i];
    if (TA == 0) {                               // (m=r2, k=half*16..+15)
      *(v8h*)&l[r2 * 32 + half * 16]     = *(const v8h*)&h[0];
      *(v8h*)&l[r2 * 32 + half * 16 + 8] = *(const v8h*)&h[8];
    } else {                                     // (k=lane, m=0..15): scatter
      #pragma unroll
      for (int m = 0; m < 16; ++m) l[m * 32 + lane] = h[m];
    }
  };
  auto stageB = [&](const float* p, _Float16* l) {
    float t4[16];
    *(float4*)&t4[0]  = ((const float4*)p)[0];
    *(float4*)&t4[4]  = ((const float4*)p)[1];
    *(float4*)&t4[8]  = ((const float4*)p)[2];
    *(float4*)&t4[12] = ((const float4*)p)[3];
    _Float16 h[16];
    #pragma unroll
    for (int i = 0; i < 16; ++i) h[i] = (_Float16)t4[i];
    if (TB == 1) {                               // (n=r2, k=half*16..+15)
      *(v8h*)&l[r2 * 32 + half * 16]     = *(const v8h*)&h[0];
      *(v8h*)&l[r2 * 32 + half * 16 + 8] = *(const v8h*)&h[8];
    } else {                                     // (k=lane, n=0..15): scatter
      #pragma unroll
      for (int n = 0; n < 16; ++n) l[n * 32 + lane] = h[n];
    }
  };
  // ---- fragments: contiguous k = 16*(lane/16) + s, same order for A and B ----
  auto fragA = [&](const _Float16* l) -> v16h {
    const int row = lane & 15;
    const int ka  = (lane < 16) ? 0 : 16;
    return *(const v16h*)&l[row * 32 + ka];          // K = ka .. ka+15
  };
  auto fragB = [&](const _Float16* l) -> v16h {
    const int row = lane & 15;
    const int kb  = (lane < 16) ? 0 : 16;
    return *(const v16h*)&l[row * 32 + kb];          // contiguous 32B
  };

  v8f acc0 = {}, acc1 = {};
  int k0 = 0;
  for (; k0 + 64 <= K; k0 += 64) {
    __builtin_prefetch(pA + 2 * stepA, 0, 0);    // speculative; safe past end
    stageA(pA, la0);          stageB(pB, lb0);
    stageA(pA + stepA, la1);  stageB(pB + stepB, lb1);
    asm volatile("s_wait_dscnt 0" ::: "memory");
    acc0 = __builtin_amdgcn_wmma_f32_16x16x32_f16(
        false, fragA(la0), false, fragB(lb0), (short)0, acc0, false, false);
    acc1 = __builtin_amdgcn_wmma_f32_16x16x32_f16(
        false, fragA(la1), false, fragB(lb1), (short)0, acc1, false, false);
    pA += 2 * stepA;
    pB += 2 * stepB;
  }
  if (k0 < K) {                                  // tail (e.g. K=32 final GEMM)
    stageA(pA, la0);  stageB(pB, lb0);
    asm volatile("s_wait_dscnt 0" ::: "memory");
    acc0 = __builtin_amdgcn_wmma_f32_16x16x32_f16(
        false, fragA(la0), false, fragB(lb0), (short)0, acc0, false, false);
  }
  #pragma unroll
  for (int j = 0; j < 8; ++j) acc0[j] += acc1[j];

  // ---- C fragment: VGPR j -> M = j + (lane>=16 ? 8 : 0), N = lane&15 ----
  const int n  = lane & 15;
  const int mb = (lane < 16) ? 0 : 8;
  float* Crow = Cb + (size_t)(mT * 16 + mb) * ldc + (nT * 16 + n);
  #pragma unroll
  for (int j = 0; j < 8; ++j) Crow[(size_t)j * ldc] = acc0[j];
}

// Per-row Gauss-Seidel sweep. Row i of u depends only on a[i,:], b, u[i,:],
// so the sequential-over-r recurrence is fully parallel across rows.
// The 32x32 Gram matrix b is broadcast into LDS via the Tensor Data Mover.
__global__ __launch_bounds__(512) void imf_sweep(
    const float* __restrict__ a, const float* __restrict__ bmat,
    float* __restrict__ u)
{
  __shared__ __align__(16) float bs[RANK * RANK];
  const int batch = blockIdx.x;
  const float* bb = bmat + (long)batch * RANK * RANK;

#if __has_builtin(__builtin_amdgcn_tensor_load_to_lds)
  if (threadIdx.x < 32) {                 // one wave issues the DMA (EXEC ignored)
    unsigned long long ga = (unsigned long long)(const void*)bb;
    unsigned int lofs = (unsigned int)(unsigned long long)(void*)&bs[0];
    u32x4 g0;                             // D# group 0 (ISA 8.3)
    g0[0] = 1u;                                        // count=1, user mode
    g0[1] = lofs;                                      // lds_addr
    g0[2] = (unsigned int)(ga & 0xffffffffu);          // global_addr[31:0]
    g0[3] = (unsigned int)((ga >> 32) & 0x01ffffffu)   // global_addr[56:32]
          | (2u << 30);                                // type=2 ("image")
    i32x8 g1;                             // D# group 1 (ISA 8.4)
    g1[0] = (int)(2u << 16);              // wg_mask=0, data_size=4B
    g1[1] = (int)(32u << 16);             // tensor_dim0[15:0] = 32 (bits 63:48)
    g1[2] = (int)(32u << 16);             // tensor_dim1[15:0] = 32 (bits 95:80)
    g1[3] = (int)(32u << 16);             // tile_dim0 = 32     (bits 127:112)
    g1[4] = 32;                           // tile_dim1 = 32
    g1[5] = 32;                           // tensor_dim0_stride = 32 elements
    g1[6] = 0; g1[7] = 0;
    i32x4 gz = {0, 0, 0, 0};              // groups 2/3 unused (2D tile)
  #if __clang_major__ >= 23
    i32x8 gz8 = {0, 0, 0, 0, 0, 0, 0, 0};
    __builtin_amdgcn_tensor_load_to_lds(g0, g1, gz, gz, gz8, 0);
  #else
    __builtin_amdgcn_tensor_load_to_lds(g0, g1, gz, gz, 0);
  #endif
    __builtin_amdgcn_s_wait_tensorcnt(0);
  }
  __syncthreads();
#else
  for (int i = threadIdx.x; i < RANK * RANK; i += blockDim.x) bs[i] = bb[i];
  __syncthreads();
#endif

  const int row = threadIdx.x;                       // 512 rows, 512 threads
  const float* ar = a + ((long)batch * NDIM + row) * RANK;
  float*       ur = u + ((long)batch * NDIM + row) * RANK;

  float uloc[RANK], aloc[RANK];
  #pragma unroll
  for (int r = 0; r < RANK; ++r) { uloc[r] = ur[r]; aloc[r] = ar[r]; }

  #pragma unroll 1
  for (int r = 0; r < RANK; ++r) {
    float brr = bs[r * RANK + r];
    float dot = 0.f;
    #pragma unroll
    for (int k = 0; k < RANK; ++k) dot = fmaf(uloc[k], bs[k * RANK + r], dot);
    float num = aloc[r] - (dot - uloc[r] * brr) + IMF_EPS;
    uloc[r] = rintf(num / (brr + IMF_EPS));          // RNE == jnp.round
  }
  #pragma unroll
  for (int r = 0; r < RANK; ++r) ur[r] = uloc[r];
}

// Deterministic projected init: u = 0, v[i][r] = round(x[r*16+8][i] * 0.1)
__global__ __launch_bounds__(512) void imf_init(
    const float* __restrict__ x, float* __restrict__ u, float* __restrict__ v)
{
  const int batch = blockIdx.x;
  const int i = threadIdx.x;
  const float* xb = x + (long)batch * NDIM * NDIM;
  float* ub = u + ((long)batch * NDIM + i) * RANK;
  float* vb = v + ((long)batch * NDIM + i) * RANK;
  #pragma unroll
  for (int r = 0; r < RANK; ++r) {
    ub[r] = 0.f;
    vb[r] = rintf(xb[(long)(r * 16 + 8) * NDIM + i] * 0.1f);
  }
}

extern "C" void kernel_launch(void* const* d_in, const int* in_sizes, int n_in,
                              void* d_out, int out_size, void* d_ws, size_t ws_size,
                              hipStream_t stream)
{
  (void)in_sizes; (void)n_in; (void)out_size; (void)ws_size;
  const float* x = (const float*)d_in[0];

  // workspace: u, v, a (each 64*512*32 f32 = 4 MB), b (64*32*32 f32)  => 12.3 MB
  float* u  = (float*)d_ws;
  float* v  = u + (long)BATCH * NDIM * RANK;
  float* a  = v + (long)BATCH * NDIM * RANK;
  float* bm = a + (long)BATCH * NDIM * RANK;
  float* out = (float*)d_out;

  const long sX = (long)NDIM * NDIM, sUV = (long)NDIM * RANK, sB = RANK * RANK;

  imf_init<<<BATCH, NDIM, 0, stream>>>(x, u, v);

  for (int it = 0; it < NUM_ITERS; ++it) {
    // a = x @ v : M=512, N=32, K=512  -> 4096 tiles, 4 waves/block
    imf_gemm<0, 0><<<(BATCH * 32 * 2) / 4, 128, 0, stream>>>(
        x, NDIM, sX,  v, RANK, sUV,  a, RANK, sUV,  NDIM, RANK, NDIM);
    // b = v^T v : M=32, N=32, K=512
    imf_gemm<1, 0><<<(BATCH * 2 * 2) / 4, 128, 0, stream>>>(
        v, RANK, sUV,  v, RANK, sUV,  bm, RANK, sB,  RANK, RANK, NDIM);
    imf_sweep<<<BATCH, NDIM, 0, stream>>>(a, bm, u);

    // a = x^T @ u : transpose staged, contiguous 64B loads per lane
    imf_gemm<1, 0><<<(BATCH * 32 * 2) / 4, 128, 0, stream>>>(
        x, NDIM, sX,  u, RANK, sUV,  a, RANK, sUV,  NDIM, RANK, NDIM);
    // b = u^T u
    imf_gemm<1, 0><<<(BATCH * 2 * 2) / 4, 128, 0, stream>>>(
        u, RANK, sUV,  u, RANK, sUV,  bm, RANK, sB,  RANK, RANK, NDIM);
    imf_sweep<<<BATCH, NDIM, 0, stream>>>(a, bm, v);
  }

  // out = u @ v^T : M=512, N=512, K=32 (single WMMA K-step per tile)
  imf_gemm<0, 1><<<(BATCH * 32 * 32) / 4, 128, 0, stream>>>(
      u, RANK, sUV,  v, RANK, sUV,  out, NDIM, sX,  NDIM, NDIM, RANK);
}
